// MoEGate_43963285242559
// MI455X (gfx1250) — matrix operand order repaired and checked by reading, not compiled, not run on hardware
//
#include <hip/hip_runtime.h>
#include <hip/hip_bf16.h>

// ---- problem constants (match reference) ----
#define TOKENS   16384
#define HIDDEN   4096
#define NEXP     256
#define TOPK     8
#define NGROUP   8
#define EPG      32          // experts per group
#define KEEPG    4           // n_group - topk_group smallest groups are KEPT (faithful)
#define SCALE    2.5f

#define TOK_TILE 32          // tokens per workgroup
#define KSTEP    32          // K per WMMA step (bf16)
#define SCP      264         // padded score stride in LDS (floats)

// LDS layout (raw block; score tile aliases the GEMM tile buffers)
#define AS_BYTES   (TOK_TILE * KSTEP * 2)        // 2 KB per A buffer
#define BS_BYTES   (NEXP * KSTEP * 2)            // 16 KB per B buffer
#define SMEM_BYTES (2 * AS_BYTES + 2 * BS_BYTES) // 36 KB total
#define OFF_AS     0
#define OFF_BS     (2 * AS_BYTES)

#if __has_builtin(__builtin_amdgcn_global_load_async_to_lds_b128) && \
    __has_builtin(__builtin_amdgcn_s_wait_asynccnt)
#define USE_ASYNC 1
#else
#define USE_ASYNC 0
#endif

typedef __attribute__((ext_vector_type(16))) __bf16 v16bf;
typedef __attribute__((ext_vector_type(8)))  float  v8f;
typedef int v4i __attribute__((vector_size(16)));   // int4 vector (builtin pointee type)

union FragU { v16bf v; uint4 q[2]; };

__device__ __forceinline__ unsigned short f2bf(float f) {
    union { float f; unsigned u; } x; x.f = f;
    unsigned r = x.u + 0x7FFFu + ((x.u >> 16) & 1u);   // round-to-nearest-even
    return (unsigned short)(r >> 16);
}

__device__ __forceinline__ unsigned ordf(float f) {
    // monotonic float->uint map (total order, smaller float -> smaller uint)
    union { float f; unsigned u; } x; x.f = f;
    return (x.u & 0x80000000u) ? ~x.u : (x.u | 0x80000000u);
}

// 16-byte global(bf16) -> LDS copy: async on CDNA5, sync fallback otherwise
__device__ __forceinline__ void copy16ToLds(const unsigned short* g, unsigned short* l) {
#if USE_ASYNC
    __builtin_amdgcn_global_load_async_to_lds_b128(
        (__attribute__((address_space(1))) v4i*)g,
        (__attribute__((address_space(3))) v4i*)l, 0, 0);
#else
    *(uint4*)l = *(const uint4*)g;
#endif
}

// ---- kernel 0: one-time W f32 -> bf16 conversion into workspace ----
__global__ __launch_bounds__(256)
void convert_w_bf16(const float* __restrict__ w, unsigned short* __restrict__ wbf) {
    const size_t i = ((size_t)blockIdx.x * 256 + threadIdx.x) * 4;
    const float4 v = *(const float4*)(w + i);
    uint2 p;
    p.x = (unsigned)f2bf(v.x) | ((unsigned)f2bf(v.y) << 16);
    p.y = (unsigned)f2bf(v.z) | ((unsigned)f2bf(v.w) << 16);
    *(uint2*)(wbf + i) = p;
}

// ---- kernel 1: fused GEMM (bf16 WMMA) + sigmoid + group-topk routing ----
__global__ __launch_bounds__(256)
void moe_gate_fused(const float* __restrict__ x,
                    const unsigned short* __restrict__ wbf,
                    const float* __restrict__ bias,
                    int* __restrict__ indsOut,
                    float* __restrict__ selOut) {
    __shared__ __align__(16) unsigned char smem[SMEM_BYTES];

    const int tid    = threadIdx.x;
    const int lane   = tid & 31;
    const int waveId = tid >> 5;              // 0..7
    const int tok0   = blockIdx.x * TOK_TILE;

    // runtime-computed LDS buffer pointers (no static pointer tables!)
    auto AsPtr = [&](int b) { return (unsigned short*)(smem + OFF_AS + b * AS_BYTES); };
    auto BsPtr = [&](int b) { return (unsigned short*)(smem + OFF_BS + b * BS_BYTES); };
    float* Sc = (float*)smem;   // aliases tile buffers; used only after the K loop

    // wave -> output tiles: mtile in {0,1} (16 tokens each), 4 consecutive N tiles
    const int mtile  = waveId & 1;
    const int ntBase = (waveId >> 1) * 4;     // N tiles ntBase..ntBase+3 (16 experts each)

    v8f acc[4] = {};

    // A-tile loader coords: each thread loads 4 consecutive f32 of x
    const int arow = tid >> 3;                // 0..31 token row
    const int acol = (tid & 7) * 4;           // 0..28 k col
    const float* xrow = x + (size_t)(tok0 + arow) * HIDDEN + acol;

    // B-tile loader: thread t owns expert row t (bf16, pre-converted)
    const unsigned short* wrow = wbf + (size_t)tid * HIDDEN;

    // fragment LDS element offsets (ushort units)
    const int aBase = (mtile * 16 + (lane & 15)) * KSTEP;
    const int ac0   = (lane < 16) ? 0 : 8;
    const int bkoff = (lane < 16) ? 0 : 8;

    const int nk = HIDDEN / KSTEP;            // 128 K-steps

    // stage A tile (f32 -> bf16, unique work per element)
    auto stageA = [&](int kb, unsigned short* As) {
        const float4 v = *(const float4*)(xrow + kb);
        uint2 p;
        p.x = (unsigned)f2bf(v.x) | ((unsigned)f2bf(v.y) << 16);
        p.y = (unsigned)f2bf(v.z) | ((unsigned)f2bf(v.w) << 16);
        *(uint2*)(&As[arow * KSTEP + acol]) = p;
    };
    // issue B tile copy (raw bf16 bytes, async on CDNA5)
    auto issueB = [&](int kb, unsigned short* Bs) {
        const unsigned short* src = wrow + kb;
        #pragma unroll
        for (int c = 0; c < 4; ++c)
            copy16ToLds(src + c * 8, &Bs[tid * KSTEP + c * 8]);
    };

    // prologue: fill buffer 0
    issueB(0, BsPtr(0));
    stageA(0, AsPtr(0));

    for (int s = 0; s < nk; ++s) {
        const int kb  = s * KSTEP;
        const int cur = s & 1;

#if USE_ASYNC
        __builtin_amdgcn_s_wait_asynccnt(0);
#endif
        __syncthreads();   // tiles for step s ready in buffer `cur`

        if (s + 1 < nk) {  // overlap next tile's movement with this step's WMMAs
            issueB(kb + KSTEP, BsPtr(cur ^ 1));
            stageA(kb + KSTEP, AsPtr(cur ^ 1));
            __builtin_prefetch(xrow + kb + 2 * KSTEP, 0, 1);  // global_prefetch_b8
        }

        // ---- A fragment (ISA 16-bit A layout) ----
        const unsigned short* As = AsPtr(cur);
        const unsigned short* Bs = BsPtr(cur);
        FragU a;
        a.q[0] = *(const uint4*)(&As[aBase + ac0]);
        a.q[1] = *(const uint4*)(&As[aBase + ac0 + 16]);

        // ---- preload all 4 B fragments, then back-to-back WMMAs ----
        FragU b[4];
        #pragma unroll
        for (int i = 0; i < 4; ++i) {
            const int bBase = ((ntBase + i) * 16 + (lane & 15)) * KSTEP;
            b[i].q[0] = *(const uint4*)(&Bs[bBase + bkoff]);
            b[i].q[1] = *(const uint4*)(&Bs[bBase + bkoff + 16]);
        }
        #pragma unroll
        for (int i = 0; i < 4; ++i)
            acc[i] = __builtin_amdgcn_wmma_f32_16x16x32_bf16(
                false, a.v, false, b[i].v, (short)0, acc[i], false, false);
    }

    __syncthreads();   // tile buffers dead; safe to alias with score tile

    // ---- epilogue: sigmoid -> LDS score tile ----
    #pragma unroll
    for (int i = 0; i < 4; ++i) {
        const int n = (ntBase + i) * 16 + (lane & 15);
        #pragma unroll
        for (int r = 0; r < 8; ++r) {
            const int m = mtile * 16 + r + ((lane >> 4) << 3);
            Sc[m * SCP + n] = __builtin_amdgcn_rcpf(1.0f + __expf(-acc[i][r]));
        }
    }
    __syncthreads();

    // ---- routing: each wave handles 4 tokens ----
    for (int t = waveId; t < TOK_TILE; t += 8) {
        // group scores: lane g<8 scans its 32 experts for top-2 sum of (sigmoid + bias)
        float gs = 0.0f;
        if (lane < NGROUP) {
            float m1 = -3.0e38f, m2 = -3.0e38f;
            const int e0 = lane * EPG;
            for (int j = 0; j < EPG; ++j) {
                const float v = Sc[t * SCP + e0 + j] + bias[e0 + j];
                if (v > m1)      { m2 = m1; m1 = v; }
                else if (v > m2) { m2 = v; }
            }
            gs = m1 + m2;
        }
        float gsArr[NGROUP];
        #pragma unroll
        for (int g = 0; g < NGROUP; ++g) gsArr[g] = __shfl(gs, g, 32);

        // keep the 4 SMALLEST groups (strict < == jax tie-break: lowest index first)
        unsigned keep = 0u;
        {
            bool used[NGROUP] = {};
            for (int r = 0; r < KEEPG; ++r) {
                float best = 3.4e38f; int bi = 0;
                for (int g = 0; g < NGROUP; ++g)
                    if (!used[g] && gsArr[g] < best) { best = gsArr[g]; bi = g; }
                used[bi] = true; keep |= (1u << bi);
            }
        }

        // per-lane candidates: slot s covers expert e = s*32 + lane (group == s)
        float sval[NGROUP];
        #pragma unroll
        for (int s = 0; s < NGROUP; ++s) {
            const int e = s * EPG + lane;
            sval[s] = ((keep >> s) & 1u) ? (Sc[t * SCP + e] + bias[e]) : 0.0f;
        }

        // 8 rounds: wave-wide min over packed (orderable_score, index) = 8 smallest of s
        unsigned selMask = 0u;
        unsigned chosen[TOPK];
        #pragma unroll
        for (int r = 0; r < TOPK; ++r) {
            unsigned long long best = ~0ull;
            #pragma unroll
            for (int s = 0; s < NGROUP; ++s) {
                if (!((selMask >> s) & 1u)) {
                    const unsigned e = (unsigned)(s * EPG + lane);
                    const unsigned long long k =
                        ((unsigned long long)ordf(sval[s]) << 32) | (unsigned long long)e;
                    if (k < best) best = k;
                }
            }
            #pragma unroll
            for (int off = 16; off >= 1; off >>= 1) {
                const unsigned long long o = __shfl_xor(best, off, 32);
                if (o < best) best = o;
            }
            const unsigned eSel = (unsigned)best;
            if ((eSel & 31u) == (unsigned)lane) selMask |= (1u << (eSel >> 5));
            chosen[r] = eSel;
        }

        // gather original sigmoid weights, normalize, scale, store
        const float wv = (lane < TOPK) ? Sc[t * SCP + chosen[lane]] : 0.0f;
        float sum = wv;
        #pragma unroll
        for (int off = 16; off >= 1; off >>= 1) sum += __shfl_xor(sum, off, 32);

        if (lane < TOPK) {
            const int gt = tok0 + t;
            indsOut[gt * TOPK + lane] = (int)chosen[lane];
            selOut [gt * TOPK + lane] = (wv / (sum + 1e-20f)) * SCALE;
        }
    }
}

extern "C" void kernel_launch(void* const* d_in, const int* in_sizes, int n_in,
                              void* d_out, int out_size, void* d_ws, size_t ws_size,
                              hipStream_t stream) {
    const float* x    = (const float*)d_in[0];
    const float* w    = (const float*)d_in[1];
    const float* bias = (const float*)d_in[2];

    int*   indsOut = (int*)d_out;                             // first output: int32 indices
    float* selOut  = ((float*)d_out) + (size_t)TOKENS * TOPK; // second output: f32 weights

    unsigned short* wbf = (unsigned short*)d_ws;              // 2 MB bf16 weights

    // kernel 0: convert W once (1M elems, 4 per thread)
    convert_w_bf16<<<dim3((NEXP * HIDDEN) / (256 * 4)), dim3(256), 0, stream>>>(w, wbf);

    // kernel 1: fused gate
    moe_gate_fused<<<dim3(TOKENS / TOK_TILE), dim3(256), 0, stream>>>(
        x, wbf, bias, indsOut, selOut);
}
